// EGNN_Ener_15728170238376
// MI455X (gfx1250) — matrix operand order repaired and checked by reading, not compiled, not run on hardware
//
#include <hip/hip_runtime.h>
#include <hip/hip_bf16.h>

// EGNN forward for MI455X (gfx1250, wave32).
// Edge MLPs done as 16-edge tiles via v_wmma_f32_16x16x32_f16.

#define HF   32   // hidden_nf
#define DXC  16   // projected coord dim
#define EIN  68   // edge MLP input dim
#define EINP 96   // padded to 3 K-chunks of 32

typedef __attribute__((ext_vector_type(16))) _Float16 v16h;
typedef __attribute__((ext_vector_type(8)))  _Float16 v8h;
typedef __attribute__((ext_vector_type(8)))  float    v8f;

// Fast SiLU: v * sigmoid(v) with single v_rcp_f32 (activation precision is fine).
__device__ __forceinline__ float silu_f(float v) {
  return v * __builtin_amdgcn_rcpf(1.0f + __expf(-v));
}

__device__ __forceinline__ v8f wmma16(v16h a, v16h b, v8f c) {
  // (neg_a, A, neg_b, B, c_mod, C, reuse_a, reuse_b)
  return __builtin_amdgcn_wmma_f32_16x16x32_f16(false, a, false, b, (short)0, c,
                                                false, false);
}

// A operand (16x32 f16) from LDS row-major tile [16][rowStride].
// Lane layout (ISA 7.12.2): lanes 0-15 row M=lane, K = kbase+{0..7,16..23};
// lanes 16-31 row M=lane-16, K = kbase+{8..15,24..31}.
__device__ __forceinline__ v16h load_a_lds(const _Float16* base, int rowStride,
                                           int kbase, int lane) {
  int m  = lane & 15;
  int ko = (lane >> 4) * 8;
  const _Float16* r = base + m * rowStride + kbase + ko;
  v8h lo = *(const v8h*)(r);
  v8h hi = *(const v8h*)(r + 16);
  v16h a;
#pragma unroll
  for (int i = 0; i < 8; ++i) { a[i] = lo[i]; a[i + 8] = hi[i]; }
  return a;
}

// B operand (32x16 f16) from global f32 weight W[out][rowLen] (row-major,
// B[k][n] = W[n][k]).  Lane layout: N = lane&15, K = kbase + 8*(lane>=16) + {0..7,16..23}.
__device__ __forceinline__ v16h load_b_gl(const float* W, int rowLen, int n,
                                          int kbase, int lane) {
  int kb = kbase + (lane >> 4) * 8;
  v16h b;
#pragma unroll
  for (int j = 0; j < 16; ++j) {
    int k = kb + j + (j >= 8 ? 8 : 0);
    float w = (k < rowLen) ? W[n * rowLen + k] : 0.0f;
    b[j] = (_Float16)w;
  }
  return b;
}

// Store SiLU(D) (16x16 f32 C-layout) into LDS f16 tile [16][HF].
// C layout: N = lane&15, M = r + 8*(lane>=16).
__device__ __forceinline__ void store_c_silu(_Float16* dst, v8f c, int tileN,
                                             int lane) {
  int mb = (lane >> 4) * 8;
  int nc = tileN * 16 + (lane & 15);
#pragma unroll
  for (int r = 0; r < 8; ++r) dst[(mb + r) * HF + nc] = (_Float16)silu_f(c[r]);
}

// ---------------------------------------------------------------------------
__global__ void egnn_init_kernel(const float* __restrict__ node_attrs,
                                 const float* __restrict__ positions,
                                 const float* __restrict__ proj_W,
                                 const float* __restrict__ emb_in_W,
                                 const float* __restrict__ emb_in_b,
                                 float* __restrict__ x, float* __restrict__ h,
                                 float* __restrict__ x_acc,
                                 float* __restrict__ m_agg,
                                 float* __restrict__ cnt, int N) {
  int n = blockIdx.x * blockDim.x + threadIdx.x;
  if (n >= N) return;
  float p0 = positions[n * 3 + 0], p1 = positions[n * 3 + 1], p2 = positions[n * 3 + 2];
#pragma unroll
  for (int d = 0; d < DXC; ++d) {
    x[n * DXC + d] = proj_W[d * 3 + 0] * p0 + proj_W[d * 3 + 1] * p1 + proj_W[d * 3 + 2] * p2;
    x_acc[n * DXC + d] = 0.0f;
  }
  float a0 = node_attrs[n * 3 + 0], a1 = node_attrs[n * 3 + 1], a2 = node_attrs[n * 3 + 2];
  for (int o = 0; o < HF; ++o) {
    h[n * HF + o] = emb_in_b[o] + emb_in_W[o * 3 + 0] * a0 +
                    emb_in_W[o * 3 + 1] * a1 + emb_in_W[o * 3 + 2] * a2;
    m_agg[n * HF + o] = 0.0f;
  }
  cnt[n] = 0.0f;
}

__global__ void egnn_degree_kernel(const int* __restrict__ ei,
                                   float* __restrict__ cnt, int E) {
  for (int e = blockIdx.x * blockDim.x + threadIdx.x; e < E;
       e += gridDim.x * blockDim.x)
    atomicAdd(&cnt[ei[e]], 1.0f);
}

// ---------------------------------------------------------------------------
// Edge kernel: one wave (32 lanes) processes 16 edges per tile via WMMA.
__global__ void __launch_bounds__(32)
egnn_edge_kernel(const float* __restrict__ x, const float* __restrict__ h,
                 const float* __restrict__ pos, const int* __restrict__ ei,
                 const float* __restrict__ W1, const float* __restrict__ b1,
                 const float* __restrict__ W2, const float* __restrict__ b2,
                 const float* __restrict__ cW1, const float* __restrict__ cb1,
                 const float* __restrict__ cW2, float* __restrict__ x_acc,
                 float* __restrict__ m_agg, int E) {
  __shared__ __attribute__((aligned(16))) _Float16 s_ein[16 * EINP];
  __shared__ __attribute__((aligned(16))) _Float16 s_hid[16 * HF];
  __shared__ __attribute__((aligned(16))) _Float16 s_m[16 * HF];
  __shared__ __attribute__((aligned(16))) _Float16 s_hc[16 * HF];
  __shared__ int   s_row[16];
  __shared__ float s_cw2[HF];

  const int lane = threadIdx.x;
  s_cw2[lane] = cW2[lane];

  // Preload per-wave weight operands in f16 B-layout (reused over all tiles).
  const int nf = lane & 15;
  v16h bW1a[3], bW1b[3];
#pragma unroll
  for (int c = 0; c < 3; ++c) {
    bW1a[c] = load_b_gl(W1, EIN, nf, c * 32, lane);
    bW1b[c] = load_b_gl(W1, EIN, 16 + nf, c * 32, lane);
  }
  v16h bW2a = load_b_gl(W2, HF, nf, 0, lane);
  v16h bW2b = load_b_gl(W2, HF, 16 + nf, 0, lane);
  v16h bC1a = load_b_gl(cW1, HF, nf, 0, lane);
  v16h bC1b = load_b_gl(cW1, HF, 16 + nf, 0, lane);
  float bs1a = b1[nf], bs1b = b1[16 + nf];
  float bs2a = b2[nf], bs2b = b2[16 + nf];
  float bsca = cb1[nf], bscb = cb1[16 + nf];

  const int numTiles = (E + 15) >> 4;
  float cd[DXC];

  for (int tile = blockIdx.x; tile < numTiles; tile += gridDim.x) {
    int myrow = -1;
    // ---- Gather: lanes 0-15 each build one edge's e_in row in LDS ----
    if (lane < 16) {
      int e = tile * 16 + lane;
      bool valid = (e < E);
      int ec = valid ? e : (E - 1);
      int r = ei[ec], cidx = ei[E + ec];
      myrow = valid ? r : -1;
      s_row[lane] = myrow;
      _Float16* row16 = s_ein + lane * EINP;
      float radial = 0.0f;
      // x rows: 4x float4 each (16B-aligned: rows are 16 floats)
      const float4* xr = (const float4*)(x + r * DXC);
      const float4* xc = (const float4*)(x + cidx * DXC);
#pragma unroll
      for (int q = 0; q < 4; ++q) {
        float4 a = xr[q], b = xc[q];
        float d0 = a.x - b.x, d1 = a.y - b.y, d2 = a.z - b.z, d3 = a.w - b.w;
        cd[q * 4 + 0] = d0; cd[q * 4 + 1] = d1;
        cd[q * 4 + 2] = d2; cd[q * 4 + 3] = d3;
        radial += d0 * d0 + d1 * d1 + d2 * d2 + d3 * d3;
      }
      // h rows: 8x float4 each (rows are 32 floats)
      const float4* hr = (const float4*)(h + r * HF);
      const float4* hc = (const float4*)(h + cidx * HF);
#pragma unroll
      for (int q = 0; q < 8; ++q) {
        float4 a = hr[q];
        row16[q * 4 + 0] = (_Float16)a.x; row16[q * 4 + 1] = (_Float16)a.y;
        row16[q * 4 + 2] = (_Float16)a.z; row16[q * 4 + 3] = (_Float16)a.w;
        float4 b = hc[q];
        row16[HF + q * 4 + 0] = (_Float16)b.x; row16[HF + q * 4 + 1] = (_Float16)b.y;
        row16[HF + q * 4 + 2] = (_Float16)b.z; row16[HF + q * 4 + 3] = (_Float16)b.w;
      }
      row16[64] = (_Float16)radial;
#pragma unroll
      for (int d = 0; d < 3; ++d)
        row16[65 + d] = (_Float16)(pos[r * 3 + d] - pos[cidx * 3 + d]);
#pragma unroll
      for (int k = EIN; k < EINP; ++k) row16[k] = (_Float16)0.0f;
    }
    __syncthreads();

    // ---- GEMM1: [16,96(pad 68)] x [96,32] ; 3 K-chunks x 2 N-tiles ----
    v8f c0, c1;
#pragma unroll
    for (int i = 0; i < 8; ++i) { c0[i] = bs1a; c1[i] = bs1b; }
#pragma unroll
    for (int c = 0; c < 3; ++c) {
      v16h a = load_a_lds(s_ein, EINP, c * 32, lane);
      c0 = wmma16(a, bW1a[c], c0);
      c1 = wmma16(a, bW1b[c], c1);
    }
    store_c_silu(s_hid, c0, 0, lane);
    store_c_silu(s_hid, c1, 1, lane);
    __syncthreads();

    // ---- GEMM2: [16,32] x [32,32] -> messages m ----
#pragma unroll
    for (int i = 0; i < 8; ++i) { c0[i] = bs2a; c1[i] = bs2b; }
    {
      v16h a = load_a_lds(s_hid, HF, 0, lane);
      c0 = wmma16(a, bW2a, c0);
      c1 = wmma16(a, bW2b, c1);
    }
    store_c_silu(s_m, c0, 0, lane);
    store_c_silu(s_m, c1, 1, lane);
    __syncthreads();

    // ---- Coord hidden: [16,32] x [32,32] ----
#pragma unroll
    for (int i = 0; i < 8; ++i) { c0[i] = bsca; c1[i] = bscb; }
    {
      v16h a = load_a_lds(s_m, HF, 0, lane);
      c0 = wmma16(a, bC1a, c0);
      c1 = wmma16(a, bC1b, c1);
    }
    store_c_silu(s_hc, c0, 0, lane);
    store_c_silu(s_hc, c1, 1, lane);
    __syncthreads();

    // ---- cm = hc . cW2 ; scatter trans into x_acc (lanes 0-15) ----
    if (lane < 16 && myrow >= 0) {
      float cm = 0.0f;
      for (int k = 0; k < HF; ++k) cm += (float)s_hc[lane * HF + k] * s_cw2[k];
#pragma unroll
      for (int d = 0; d < DXC; ++d)
        atomicAdd(&x_acc[myrow * DXC + d], cd[d] * cm);
    }
    // ---- scatter m into m_agg (all 32 lanes: edge=lane&15, 16 feats each) ----
    {
      int el = lane & 15;
      int r = s_row[el];
      if (r >= 0) {
        int fo = (lane >> 4) * 16;
        for (int f = 0; f < 16; ++f)
          atomicAdd(&m_agg[r * HF + fo + f], (float)s_m[el * HF + fo + f]);
      }
    }
    __syncthreads();  // protect LDS before next tile overwrites
  }
}

// ---------------------------------------------------------------------------
// Node update: x += x_acc/max(cnt,1); h += MLP([h, m_agg]); zero accumulators.
__global__ void egnn_node_kernel(float* __restrict__ x, float* __restrict__ h,
                                 float* __restrict__ x_acc,
                                 float* __restrict__ m_agg,
                                 const float* __restrict__ cnt,
                                 const float* __restrict__ nW1,
                                 const float* __restrict__ nb1,
                                 const float* __restrict__ nW2,
                                 const float* __restrict__ nb2, int N) {
  __shared__ float sW1[HF * 2 * HF];  // [32][64]
  __shared__ float sW2[HF * HF];      // [32][32]
  int tid = threadIdx.x;
  for (int i = tid; i < HF * 2 * HF; i += blockDim.x) sW1[i] = nW1[i];
  for (int i = tid; i < HF * HF; i += blockDim.x) sW2[i] = nW2[i];
  __syncthreads();

  for (int n = blockIdx.x * blockDim.x + tid; n < N;
       n += gridDim.x * blockDim.x) {
    float inv = __builtin_amdgcn_rcpf(fmaxf(cnt[n], 1.0f));
#pragma unroll
    for (int d = 0; d < DXC; ++d) {
      x[n * DXC + d] += x_acc[n * DXC + d] * inv;
      x_acc[n * DXC + d] = 0.0f;
    }
    float hv[HF], mv[HF], hid[HF];
    for (int k = 0; k < HF; ++k) {
      hv[k] = h[n * HF + k];
      mv[k] = m_agg[n * HF + k];
      m_agg[n * HF + k] = 0.0f;
    }
    for (int o = 0; o < HF; ++o) {
      float s = nb1[o];
      for (int k = 0; k < HF; ++k) s += sW1[o * 2 * HF + k] * hv[k];
      for (int k = 0; k < HF; ++k) s += sW1[o * 2 * HF + HF + k] * mv[k];
      hid[o] = silu_f(s);
    }
    for (int o = 0; o < HF; ++o) {
      float s = nb2[o];
      for (int k = 0; k < HF; ++k) s += sW2[o * HF + k] * hid[k];
      h[n * HF + o] = hv[o] + s;
    }
  }
}

__global__ void egnn_final_kernel(const float* __restrict__ x,
                                  const float* __restrict__ lin_W,
                                  float* __restrict__ out, int N) {
  int n = blockIdx.x * blockDim.x + threadIdx.x;
  if (n >= N) return;
#pragma unroll
  for (int j = 0; j < 3; ++j) {
    float s = 0.0f;
#pragma unroll
    for (int d = 0; d < DXC; ++d) s += x[n * DXC + d] * lin_W[j * DXC + d];
    out[n * 3 + j] = s;
  }
}

// ---------------------------------------------------------------------------
extern "C" void kernel_launch(void* const* d_in, const int* in_sizes, int n_in,
                              void* d_out, int out_size, void* d_ws,
                              size_t ws_size, hipStream_t stream) {
  const float* node_attrs = (const float*)d_in[0];
  const float* positions  = (const float*)d_in[1];
  const int*   edge_index = (const int*)d_in[2];
  const float* proj_W     = (const float*)d_in[3];
  const float* emb_in_W   = (const float*)d_in[4];
  const float* emb_in_b   = (const float*)d_in[5];
  const float* edge_W1    = (const float*)d_in[6];
  const float* edge_b1    = (const float*)d_in[7];
  const float* edge_W2    = (const float*)d_in[8];
  const float* edge_b2    = (const float*)d_in[9];
  const float* node_W1    = (const float*)d_in[10];
  const float* node_b1    = (const float*)d_in[11];
  const float* node_W2    = (const float*)d_in[12];
  const float* node_b2    = (const float*)d_in[13];
  const float* coord_W1   = (const float*)d_in[14];
  const float* coord_b1   = (const float*)d_in[15];
  const float* coord_W2   = (const float*)d_in[16];
  const float* lin_W      = (const float*)d_in[19];

  const int N = in_sizes[0] / 3;
  const int E = in_sizes[2] / 2;
  const int L = in_sizes[7] / HF;  // edge_b1 is [L,32]

  // Workspace layout (f32): x[N*16] | h[N*32] | x_acc[N*16] | m_agg[N*32] | cnt[N]
  float* ws    = (float*)d_ws;
  float* x     = ws;
  float* h     = x + (size_t)N * DXC;
  float* x_acc = h + (size_t)N * HF;
  float* m_agg = x_acc + (size_t)N * DXC;
  float* cnt   = m_agg + (size_t)N * HF;
  (void)ws_size; (void)n_in; (void)out_size;

  egnn_init_kernel<<<(N + 255) / 256, 256, 0, stream>>>(
      node_attrs, positions, proj_W, emb_in_W, emb_in_b, x, h, x_acc, m_agg,
      cnt, N);
  egnn_degree_kernel<<<2048, 256, 0, stream>>>(edge_index, cnt, E);

  for (int l = 0; l < L; ++l) {
    egnn_edge_kernel<<<8192, 32, 0, stream>>>(
        x, h, positions, edge_index,
        edge_W1 + (size_t)l * HF * EIN, edge_b1 + l * HF,
        edge_W2 + (size_t)l * HF * HF, edge_b2 + l * HF,
        coord_W1 + (size_t)l * HF * HF, coord_b1 + l * HF,
        coord_W2 + l * HF, x_acc, m_agg, E);
    egnn_node_kernel<<<1024, 256, 0, stream>>>(
        x, h, x_acc, m_agg, cnt,
        node_W1 + (size_t)l * HF * 2 * HF, node_b1 + l * HF,
        node_W2 + (size_t)l * HF * HF, node_b2 + l * HF, N);
  }

  egnn_final_kernel<<<(N + 255) / 256, 256, 0, stream>>>(x, lin_W,
                                                         (float*)d_out, N);
}
